// Model_77773267796094
// MI455X (gfx1250) — matrix operand order repaired and checked by reading, not compiled, not run on hardware
//
#include <hip/hip_runtime.h>
#include <hip/hip_bf16.h>

typedef __attribute__((ext_vector_type(16))) __bf16 v16bf;
typedef __attribute__((ext_vector_type(8)))  float  v8f;

#define B_   16384
#define T_   80
#define D_   12
#define H_   128
#define ROWS 16      // batch rows per workgroup
#define NTH  512     // 16 waves (wave32)

static __device__ __forceinline__ __bf16 tobf(float f) { return (__bf16)f; }

static __device__ __forceinline__ float sigm(float x) {
  return __builtin_amdgcn_rcpf(1.0f + __expf(-x));
}
static __device__ __forceinline__ float tanh_fast(float x) {
  const float ax = fabsf(x);
  const float t  = __expf(-2.0f * ax);
  const float r  = (1.0f - t) * __builtin_amdgcn_rcpf(1.0f + t);
  return copysignf(r, x);
}

// Pre-pass: denom[t] = sum over [B,D] of masks[:,t,:] + 1e-5
__global__ void brits_prep(const float* __restrict__ masks, float* __restrict__ denom) {
  __shared__ float red[256];
  const int t = blockIdx.x, tid = threadIdx.x;
  float s = 0.f;
  for (int b = tid; b < B_; b += 256) {
    const float* p = masks + ((size_t)b * T_ + t) * D_;
#pragma unroll
    for (int c = 0; c < D_; ++c) s += p[c];
  }
  red[tid] = s;
  __syncthreads();
  for (int off = 128; off > 0; off >>= 1) {
    if (tid < off) red[tid] += red[tid + off];
    __syncthreads();
  }
  if (tid == 0) denom[t] = red[0] + 1e-5f;
}

__global__ __launch_bounds__(NTH)
void brits_main(const float* __restrict__ values, const float* __restrict__ masks,
                const float* __restrict__ deltas,
                const float* __restrict__ td_h_W, const float* __restrict__ td_h_b,
                const float* __restrict__ td_x_W, const float* __restrict__ td_x_b,
                const float* __restrict__ hist_W, const float* __restrict__ hist_b,
                const float* __restrict__ feat_W, const float* __restrict__ feat_b,
                const float* __restrict__ wc_W,  const float* __restrict__ wc_b,
                const float* __restrict__ W_ih,  const float* __restrict__ b_ih,
                const float* __restrict__ W_hh,  const float* __restrict__ b_hh,
                const float* __restrict__ out_W, const float* __restrict__ out_b,
                const float* __restrict__ denom,
                float* __restrict__ loss_partials, float* __restrict__ preds,
                float* __restrict__ imps) {
  __shared__ float sh_h[ROWS][H_];
  __shared__ float sh_c[ROWS][H_];
  __shared__ float sh_g[ROWS][4 * H_];
  __shared__ float sh_x[ROWS][D_], sh_m[ROWS][D_], sh_d[ROWS][D_];
  __shared__ float sh_xh[ROWS][D_], sh_xc[ROWS][D_], sh_gx[ROWS][D_], sh_cc[ROWS][D_];
  __shared__ float sw_featW[D_][D_];
  __shared__ float sw_featb[D_];
  __shared__ float sw_wcW[D_][2 * D_];
  __shared__ float sw_wcb[D_];
  __shared__ float sw_diag[D_], sw_tdxb[D_];
  __shared__ float sw_outW[H_];
  __shared__ v16bf sw_BhistF[4][32];  // pre-built per-lane B fragments of hist_W
  __shared__ float sred[NTH];

  const int tid = threadIdx.x;
  const int wave = tid >> 5, lane = tid & 31;
  const int laneHi = lane >> 4, laneLo = lane & 15;
  const int wv = __builtin_amdgcn_readfirstlane(wave);  // wave-uniform (SGPR)
  const int wgB = blockIdx.x * ROWS;

  // zero h/c state; stage small weights in LDS
  for (int e = tid; e < ROWS * H_; e += NTH) {
    (&sh_h[0][0])[e] = 0.f;
    (&sh_c[0][0])[e] = 0.f;
  }
  for (int e = tid; e < H_; e += NTH) sw_outW[e] = out_W[e];
  for (int e = tid; e < D_ * D_; e += NTH) {
    int r = e / D_, c = e % D_;
    sw_featW[r][c] = (r == c) ? 0.f : feat_W[e];  // zero-diagonal
  }
  for (int e = tid; e < D_ * 2 * D_; e += NTH) (&sw_wcW[0][0])[e] = wc_W[e];
  if (tid < D_) {
    sw_featb[tid] = feat_b[tid];
    sw_wcb[tid]   = wc_b[tid];
    sw_diag[tid]  = td_x_W[tid * D_ + tid];  // diagonal of td_x_W
    sw_tdxb[tid]  = td_x_b[tid];
  }
  // wave 0 pre-builds hist_W B-fragments into LDS (only wave 0 consumes them)
  if (tid < 32) {
#pragma unroll
    for (int kt = 0; kt < 4; ++kt) {
      v16bf bh;
#pragma unroll
      for (int e = 0; e < 16; ++e) {
        const int v = e >> 1, half = e & 1;
        const int K = kt * 32 + laneHi * 16 + 2 * v + half;
        bh[e] = (laneLo < D_) ? tobf(hist_W[laneLo * H_ + K]) : tobf(0.f);
      }
      sw_BhistF[kt][lane] = bh;
    }
  }

  // ---- Register-resident bf16 B-fragments ----
  // B 32x16 k order used here (contiguous): lanes 0-15 hold K=0..15, lanes 16-31 K=16..31;
  // VGPR v holds K=2v,2v+1. N = laneLo.
  v16bf Bhh[2][4];  // W_hh  [512,128] -> 2 N-tiles/wave x 4 K-steps
  v16bf Bih[2];     // W_ih  [512, 24] -> K padded to 32
  v16bf Btdh;       // td_h_W [128,12] as B: wave (mod 8) owns N-tile, K=12 pad 32
  float biasv[2], tdhb_r, histb_r;
#pragma unroll
  for (int j = 0; j < 2; ++j) {
    const int col = (wave * 2 + j) * 16 + laneLo;  // global gate column (N)
    biasv[j] = b_ih[col] + b_hh[col];
#pragma unroll
    for (int kt = 0; kt < 4; ++kt) {
#pragma unroll
      for (int e = 0; e < 16; ++e) {
        const int v = e >> 1, half = e & 1;
        const int K = kt * 32 + laneHi * 16 + 2 * v + half;
        Bhh[j][kt][e] = tobf(W_hh[col * H_ + K]);
      }
    }
#pragma unroll
    for (int e = 0; e < 16; ++e) {
      const int v = e >> 1, half = e & 1;
      const int K = laneHi * 16 + 2 * v + half;
      Bih[j][e] = (K < 2 * D_) ? tobf(W_ih[col * (2 * D_) + K]) : tobf(0.f);
    }
  }
  {
    const int col8 = (wave & 7) * 16 + laneLo;  // gamma_h column for this wave
    tdhb_r = td_h_b[col8];
#pragma unroll
    for (int e = 0; e < 16; ++e) {
      const int v = e >> 1, half = e & 1;
      const int K = laneHi * 16 + 2 * v + half;
      Btdh[e] = (K < D_) ? tobf(td_h_W[col8 * D_ + K]) : tobf(0.f);
    }
    histb_r = (laneLo < D_) ? hist_b[laneLo] : 0.f;
  }
  __syncthreads();

  float lossAcc = 0.f;
  const int rr  = (tid < ROWS * D_) ? (tid / D_) : 0;
  const int cc_ = (tid < ROWS * D_) ? (tid % D_) : 0;

  for (int t = 0; t < T_; ++t) {
    const float invden = 1.0f / denom[t];

    // P0: load x, m, d tile (streamed once -> non-temporal)
    if (tid < ROWS * D_) {
      const size_t gidx = ((size_t)(wgB + rr) * T_ + t) * D_ + cc_;
      sh_x[rr][cc_] = __builtin_nontemporal_load(values + gidx);
      sh_m[rr][cc_] = __builtin_nontemporal_load(masks + gidx);
      sh_d[rr][cc_] = __builtin_nontemporal_load(deltas + gidx);
      if (t + 1 < T_) {  // global_prefetch_b8 next step
        __builtin_prefetch(values + gidx + D_, 0, 0);
        __builtin_prefetch(masks + gidx + D_, 0, 0);
        __builtin_prefetch(deltas + gidx + D_, 0, 0);
      }
    }
    __syncthreads();

    // P1: gamma_h = exp(-relu(d @ td_h_W.T + b)) via WMMA; h *= gamma_h.
    // Wave-uniform scalar branch: only waves 0-7 execute (EXEC all-1s inside).
    if (wv < 8) {
      // A 16-bit 16x32 layout (matches B): lane<16 -> K 0..15; lane>=16 -> K 16..31
      v16bf Ad;
#pragma unroll
      for (int e = 0; e < 16; ++e) {
        const int v = e >> 1, half = e & 1;
        const int K = laneHi * 16 + 2 * v + half;
        Ad[e] = (K < D_) ? tobf(sh_d[laneLo][K]) : tobf(0.f);
      }
      v8f acc;
#pragma unroll
      for (int v = 0; v < 8; ++v) acc[v] = tdhb_r;
      acc = __builtin_amdgcn_wmma_f32_16x16x32_bf16(false, Ad, false, Btdh,
                                                    (short)0, acc, false, false);
      const int col = wave * 16 + laneLo;
#pragma unroll
      for (int v = 0; v < 8; ++v) {
        const float g = __expf(-fmaxf(acc[v], 0.f));
        sh_h[v + 8 * laneHi][col] *= g;
      }
    }
    __syncthreads();

    // Build A fragments of decayed h once; reused by x_h GEMM and gate GEMM.
    v16bf Ah[4];
#pragma unroll
    for (int kt = 0; kt < 4; ++kt) {
#pragma unroll
      for (int e = 0; e < 16; ++e) {
        const int v = e >> 1, half = e & 1;
        const int K = kt * 32 + laneHi * 16 + 2 * v + half;
        Ah[kt][e] = tobf(sh_h[laneLo][K]);
      }
    }

    // P2-wmma: x_h = h @ hist_W.T + b (N=12 padded to 16); wave 0 only.
    if (wv == 0) {
      v8f acc;
#pragma unroll
      for (int v = 0; v < 8; ++v) acc[v] = histb_r;
#pragma unroll
      for (int kt = 0; kt < 4; ++kt) {
        const v16bf bh = sw_BhistF[kt][lane];
        acc = __builtin_amdgcn_wmma_f32_16x16x32_bf16(false, Ah[kt], false, bh,
                                                      (short)0, acc, false, false);
      }
      if (laneLo < D_) {
#pragma unroll
        for (int v = 0; v < 8; ++v) sh_xh[v + 8 * laneHi][laneLo] = acc[v];
      }
    }
    __syncthreads();

    // P2e: loss1 ; x_c ; gamma_x
    if (tid < ROWS * D_) {
      const float s = sh_xh[rr][cc_];
      const float x = sh_x[rr][cc_], mm = sh_m[rr][cc_];
      lossAcc += fabsf(x - s) * mm * invden;
      sh_xc[rr][cc_] = mm * x + (1.f - mm) * s;
      const float dd = sh_d[rr][cc_];
      sh_gx[rr][cc_] = __expf(-fmaxf(dd * sw_diag[cc_] + sw_tdxb[cc_], 0.f));
    }
    __syncthreads();

    // P3: z_h, alpha, c_h, c_c, loss2+3, store imputation (NT stream out)
    if (tid < ROWS * D_) {
      const float x = sh_x[rr][cc_], mm = sh_m[rr][cc_];
      float z = sw_featb[cc_];
#pragma unroll
      for (int k = 0; k < D_; ++k) z += sh_xc[rr][k] * sw_featW[cc_][k];
      lossAcc += fabsf(x - z) * mm * invden;
      float a = sw_wcb[cc_];
#pragma unroll
      for (int k = 0; k < D_; ++k)
        a += sh_gx[rr][k] * sw_wcW[cc_][k] + sh_m[rr][k] * sw_wcW[cc_][D_ + k];
      const float chv = a * z + (1.f - a) * sh_xh[rr][cc_];
      lossAcc += fabsf(x - chv) * mm * invden;
      const float ccv = mm * x + (1.f - mm) * chv;
      sh_cc[rr][cc_] = ccv;
      __builtin_nontemporal_store(ccv, imps + ((size_t)(wgB + rr) * T_ + t) * D_ + cc_);
    }
    __syncthreads();

    // P4: gates = [c_c,m]@W_ih.T + h@W_hh.T + b via BF16 WMMA (Ah reused)
    {
      v16bf Aih;
#pragma unroll
      for (int e = 0; e < 16; ++e) {
        const int v = e >> 1, half = e & 1;
        const int K = laneHi * 16 + 2 * v + half;
        const float val = (K < D_) ? sh_cc[laneLo][K]
                                   : ((K < 2 * D_) ? sh_m[laneLo][K - D_] : 0.f);
        Aih[e] = tobf(val);
      }
#pragma unroll
      for (int j = 0; j < 2; ++j) {
        v8f acc;
#pragma unroll
        for (int v = 0; v < 8; ++v) acc[v] = biasv[j];
        acc = __builtin_amdgcn_wmma_f32_16x16x32_bf16(false, Aih, false, Bih[j],
                                                      (short)0, acc, false, false);
#pragma unroll
        for (int kt = 0; kt < 4; ++kt)
          acc = __builtin_amdgcn_wmma_f32_16x16x32_bf16(false, Ah[kt], false, Bhh[j][kt],
                                                        (short)0, acc, false, false);
        const int ncol = (wave * 2 + j) * 16 + laneLo;
#pragma unroll
        for (int v = 0; v < 8; ++v) sh_g[v + 8 * laneHi][ncol] = acc[v];
      }
    }
    __syncthreads();

    // P5: LSTM pointwise update (branch-free fast activations)
#pragma unroll
    for (int it = 0; it < (ROWS * H_) / NTH; ++it) {
      const int e = tid + it * NTH;
      const int r = e >> 7, hc = e & (H_ - 1);
      const float ig = sh_g[r][hc],          fg = sh_g[r][H_ + hc];
      const float gg = sh_g[r][2 * H_ + hc], og = sh_g[r][3 * H_ + hc];
      const float cn = sigm(fg) * sh_c[r][hc] + sigm(ig) * tanh_fast(gg);
      sh_c[r][hc] = cn;
      sh_h[r][hc] = sigm(og) * tanh_fast(cn);
    }
    __syncthreads();
  }

  // predictions = sigmoid(h @ out_W.T + out_b)
  if (tid < ROWS) {
    float s = out_b[0];
#pragma unroll 8
    for (int k = 0; k < H_; ++k) s += sh_h[tid][k] * sw_outW[k];
    preds[wgB + tid] = sigm(s);
  }

  // deterministic per-WG loss partial
  sred[tid] = lossAcc;
  __syncthreads();
  for (int off = NTH / 2; off > 0; off >>= 1) {
    if (tid < off) sred[tid] += sred[tid + off];
    __syncthreads();
  }
  if (tid == 0) loss_partials[blockIdx.x] = sred[0];
}

__global__ void brits_loss_reduce(const float* __restrict__ partials,
                                  float* __restrict__ loss_out) {
  __shared__ float red[256];
  const int tid = threadIdx.x;
  float s = 0.f;
  for (int i = tid; i < B_ / ROWS; i += 256) s += partials[i];
  red[tid] = s;
  __syncthreads();
  for (int off = 128; off > 0; off >>= 1) {
    if (tid < off) red[tid] += red[tid + off];
    __syncthreads();
  }
  if (tid == 0) loss_out[0] = red[0] * (1.0f / T_);
}

extern "C" void kernel_launch(void* const* d_in, const int* in_sizes, int n_in,
                              void* d_out, int out_size, void* d_ws, size_t ws_size,
                              hipStream_t stream) {
  (void)in_sizes; (void)n_in; (void)out_size; (void)ws_size;
  const float* values = (const float*)d_in[0];
  const float* masks  = (const float*)d_in[1];
  const float* deltas = (const float*)d_in[2];
  const float* td_h_W = (const float*)d_in[3];
  const float* td_h_b = (const float*)d_in[4];
  const float* td_x_W = (const float*)d_in[5];
  const float* td_x_b = (const float*)d_in[6];
  const float* hist_W = (const float*)d_in[7];
  const float* hist_b = (const float*)d_in[8];
  const float* feat_W = (const float*)d_in[9];
  const float* feat_b = (const float*)d_in[10];
  const float* wc_W   = (const float*)d_in[11];
  const float* wc_b   = (const float*)d_in[12];
  const float* W_ih   = (const float*)d_in[13];
  const float* b_ih   = (const float*)d_in[14];
  const float* W_hh   = (const float*)d_in[15];
  const float* b_hh   = (const float*)d_in[16];
  const float* out_W  = (const float*)d_in[17];
  const float* out_b  = (const float*)d_in[18];

  float* out   = (float*)d_out;
  float* lossp = out;           // [1]
  float* preds = out + 1;       // [B]
  float* imps  = out + 1 + B_;  // [B,T,D]

  float* denom         = (float*)d_ws;         // [T]
  float* loss_partials = denom + T_;           // [B/ROWS]

  brits_prep<<<T_, 256, 0, stream>>>(masks, denom);
  brits_main<<<B_ / ROWS, NTH, 0, stream>>>(
      values, masks, deltas, td_h_W, td_h_b, td_x_W, td_x_b, hist_W, hist_b,
      feat_W, feat_b, wc_W, wc_b, W_ih, b_ih, W_hh, b_hh, out_W, out_b,
      denom, loss_partials, preds, imps);
  brits_loss_reduce<<<1, 256, 0, stream>>>(loss_partials, lossp);
}
